// GConvEnSparse_64828236365870
// MI455X (gfx1250) — compile-verified, matrix operand order checked by reading, not run on hardware
//
#include <hip/hip_runtime.h>
#include <hip/hip_bf16.h>

// ---------------------------------------------------------------------------
// EGNN layer for MI455X (gfx1250): bf16 WMMA (f32 accum) for all GEMMs.
// Edge kernel: 32 edges/block, B-fragments shared across 2 M-tiles to halve
// L2 traffic on the dominant 257x514 GEMM.
// ---------------------------------------------------------------------------

typedef __bf16 bf16_t;
typedef __attribute__((ext_vector_type(16))) __bf16 v16bf;
typedef __attribute__((ext_vector_type(8)))  __bf16 v8bf;
typedef __attribute__((ext_vector_type(8)))  float  v8f;

#define NN 50000
#define EE 500000
// GEMM shapes (K real, K tiles of 32, N real, N tiles of 16)
#define K1 257
#define KT1 9
#define N1 514
#define NT1 33
#define K2 514
#define KT2 17
#define N2 64
#define NT2 4
#define K3 64
#define KT3 2
#define N3 256
#define NT3 16
#define K4 256
#define KT4 8
#define N4 1
#define NT4 1
#define KN1 192
#define KTN1 6
#define NN1 256
#define NTN1 16
#define KN2 256
#define KTN2 8
#define NN2 128
#define NTN2 8

__device__ __forceinline__ bf16_t f2bf(float f) {
  unsigned u = __builtin_bit_cast(unsigned, f);
  unsigned r = u + 0x7FFFu + ((u >> 16) & 1u); // round-to-nearest-even
  unsigned short h = (unsigned short)(r >> 16);
  return __builtin_bit_cast(bf16_t, h);
}

__device__ __forceinline__ float bf2f(bf16_t b) {
  unsigned short h = __builtin_bit_cast(unsigned short, b);
  unsigned u = (unsigned)h << 16;
  return __builtin_bit_cast(float, u);
}

__device__ __forceinline__ float silu_f(float x) {
  return x / (1.0f + __expf(-x));
}

// A-fragment for v_wmma_f32_16x16x32_bf16 from row-major bf16 LDS tile.
// ISA layout: lane L -> row m = L&15, K-half kh = L>>4;
// elements 0..7  cover K = kbase + 8*kh + [0..7]     (contiguous, 16B)
// elements 8..15 cover K = kbase + 16 + 8*kh + [0..7] (contiguous, 16B)
__device__ __forceinline__ v16bf load_a_frag(const bf16_t* A, int stride,
                                             int lane, int kbase) {
  int m = lane & 15;
  int kh = lane >> 4;
  const bf16_t* p = A + m * stride + kbase + 8 * kh;
  v8bf lo = *(const v8bf*)(p);
  v8bf hi = *(const v8bf*)(p + 16);
  return __builtin_shufflevector(lo, hi, 0, 1, 2, 3, 4, 5, 6, 7,
                                 8, 9, 10, 11, 12, 13, 14, 15);
}

// B fragments are pre-packed lane-major by pack_weight_kernel:
// tile (kt,nt): 32 lanes x 16 bf16 contiguous -> one 32B load per lane.
__device__ __forceinline__ v16bf load_b_frag(const bf16_t* B, int NT,
                                             int kt, int nt, int lane) {
  return *(const v16bf*)(B + ((long long)(kt * NT + nt) * 32 + lane) * 16);
}

__device__ __forceinline__ v8f wmma_bf16(v16bf a, v16bf b, v8f c) {
  return __builtin_amdgcn_wmma_f32_16x16x32_bf16(
      /*neg_a=*/false, a, /*neg_b=*/false, b,
      /*c_mod=*/(short)0, c, /*reuse_a=*/false, /*reuse_b=*/false);
}

// ---------------------------------------------------------------------------
// Prep kernels
// ---------------------------------------------------------------------------

__global__ __launch_bounds__(256) void zero_f32_kernel(float* p, long long n) {
  long long i = (long long)blockIdx.x * 256 + threadIdx.x;
  if (i < n) p[i] = 0.0f;
}

// Pack row-major f32 weight [K x N] into WMMA-B fragment-major bf16 layout,
// zero-padded to KT*32 x NT*16.
__global__ __launch_bounds__(256) void pack_weight_kernel(
    const float* __restrict__ W, bf16_t* __restrict__ dst,
    int K, int N, int KT, int NT) {
  int idx = blockIdx.x * 256 + threadIdx.x;
  int total = KT * NT * 32;
  if (idx >= total) return;
  int lane = idx & 31;
  int tile = idx >> 5;
  int nt = tile % NT;
  int kt = tile / NT;
  int n = nt * 16 + (lane & 15);
  int kh = lane >> 4;
  bf16_t* o = dst + (long long)idx * 16;
#pragma unroll
  for (int e = 0; e < 16; ++e) {
    int k = kt * 32 + 8 * kh + e + ((e >= 8) ? 8 : 0);
    float v = (k < K && n < N) ? W[(long long)k * N + n] : 0.0f;
    o[e] = f2bf(v);
  }
}

// ---------------------------------------------------------------------------
// Edge kernel: 32 edges / block (2 M-tiles), 256 threads (8 waves).
// LDS overlay:
//   [0      , 18432)  sA (32x288 bf16), reused as sT (32x256 bf16) in GEMM3/4
//   [18432  , 53248)  sH (32x544 bf16)
//   [53248  , 57344)  sM (32x64 bf16)  m_ij
//   tail: sRel, sCw, sDst
// ---------------------------------------------------------------------------

#define SMEM_BYTES (18432 + 34816 + 4096)

__global__ __launch_bounds__(256) void egnn_edge_kernel(
    const float* __restrict__ x, const int* __restrict__ ei,
    const float* __restrict__ b_e1, const float* __restrict__ b_e2,
    const float* __restrict__ b_c1, const float* __restrict__ b_c2,
    const bf16_t* __restrict__ We1, const bf16_t* __restrict__ We2,
    const bf16_t* __restrict__ Wc1, const bf16_t* __restrict__ Wc2,
    float* __restrict__ m_i, float* __restrict__ w_i,
    float* __restrict__ r_i) {
  __shared__ alignas(16) char smem[SMEM_BYTES];
  __shared__ float sRel[32][3];
  __shared__ float sCw[32];
  __shared__ int   sDst[32];

  bf16_t* sA = (bf16_t*)smem;               // 32x288
  bf16_t* sT = (bf16_t*)smem;               // 32x256 (aliases sA, GEMM3+)
  bf16_t* sH = (bf16_t*)(smem + 18432);     // 32x544
  bf16_t* sM = (bf16_t*)(smem + 53248);     // 32x64

  const int tid = threadIdx.x;
  const int lane = tid & 31;
  const int wave = tid >> 5;
  const int e0 = blockIdx.x * 32;

  // ---- Phase 0: gather A = [node[dst] | node[src] | rel_dist | 0] ----
  {
    int m = tid >> 3;   // edge slot 0..31
    int sub = tid & 7;  // 8 threads per edge
    int e = e0 + m;
    int s = ei[e];
    int d = ei[EE + e];
    const float* xs = x + (long long)s * 131;
    const float* xd = x + (long long)d * 131;
#pragma unroll
    for (int j = 0; j < 16; ++j) {
      int k = sub * 16 + j;
      sA[m * 288 + k]       = f2bf(xd[k]); // node[dst]
      sA[m * 288 + 128 + k] = f2bf(xs[k]); // node[src]
    }
    if (sub == 0) {
      float r0 = xs[128] - xd[128];
      float r1 = xs[129] - xd[129];
      float r2 = xs[130] - xd[130];
      float dist = sqrtf(r0 * r0 + r1 * r1 + r2 * r2);
      sA[m * 288 + 256] = f2bf(dist);
#pragma unroll
      for (int k = 257; k < 288; ++k) sA[m * 288 + k] = f2bf(0.0f);
      sRel[m][0] = r0; sRel[m][1] = r1; sRel[m][2] = r2;
      sDst[m] = d;
    }
    // zero sH pad columns 528..543 for all 32 rows (512 slots, 2/thread)
#pragma unroll
    for (int i = 0; i < 2; ++i) {
      int idx = tid + i * 256;
      sH[(idx >> 4) * 544 + 528 + (idx & 15)] = f2bf(0.0f);
    }
  }
  __syncthreads();

  // ---- GEMM1: (32x288) @ We1 -> silu -> sH (32x544) ----
  // B fragment shared across both M-tiles: halves L2 B traffic.
  {
    for (int nt = wave; nt < NT1; nt += 8) {
      int n = nt * 16 + (lane & 15);
      float bias = (n < N1) ? b_e1[n] : 0.0f;
      v8f acc0, acc1;
#pragma unroll
      for (int r = 0; r < 8; ++r) { acc0[r] = bias; acc1[r] = bias; }
#pragma unroll
      for (int kt = 0; kt < KT1; ++kt) {
        v16bf b  = load_b_frag(We1, NT1, kt, nt, lane);
        v16bf a0 = load_a_frag(sA,            288, lane, kt * 32);
        v16bf a1 = load_a_frag(sA + 16 * 288, 288, lane, kt * 32);
        acc0 = wmma_bf16(a0, b, acc0);
        acc1 = wmma_bf16(a1, b, acc1);
      }
#pragma unroll
      for (int r = 0; r < 8; ++r) {
        int m = r + 8 * (lane >> 4);
        int col = nt * 16 + (lane & 15);
        sH[m * 544 + col]        = f2bf(silu_f(acc0[r]));
        sH[(m + 16) * 544 + col] = f2bf(silu_f(acc1[r]));
      }
    }
  }
  __syncthreads();

  // ---- GEMM2: (32x544) @ We2 -> silu -> m_ij (32x64), 8 tasks/8 waves ----
  {
    int mt = wave >> 2;  // M-tile 0..1
    int nt = wave & 3;   // N-tile 0..3
    int n = nt * 16 + (lane & 15);
    float bias = b_e2[n];
    v8f acc;
#pragma unroll
    for (int r = 0; r < 8; ++r) acc[r] = bias;
#pragma unroll
    for (int kt = 0; kt < KT2; ++kt) {
      v16bf a = load_a_frag(sH + mt * 16 * 544, 544, lane, kt * 32);
      acc = wmma_bf16(a, load_b_frag(We2, NT2, kt, nt, lane), acc);
    }
#pragma unroll
    for (int r = 0; r < 8; ++r) {
      int m = mt * 16 + r + 8 * (lane >> 4);
      sM[m * 64 + n] = f2bf(silu_f(acc[r]));
    }
  }
  __syncthreads();

  // ---- GEMM3: (32x64) @ Wc1 -> silu -> sT (32x256), B shared over M ----
  {
#pragma unroll
    for (int i = 0; i < 2; ++i) {
      int nt = wave + 8 * i;
      float bias = b_c1[nt * 16 + (lane & 15)];
      v8f acc0, acc1;
#pragma unroll
      for (int r = 0; r < 8; ++r) { acc0[r] = bias; acc1[r] = bias; }
#pragma unroll
      for (int kt = 0; kt < KT3; ++kt) {
        v16bf b  = load_b_frag(Wc1, NT3, kt, nt, lane);
        v16bf a0 = load_a_frag(sM,           64, lane, kt * 32);
        v16bf a1 = load_a_frag(sM + 16 * 64, 64, lane, kt * 32);
        acc0 = wmma_bf16(a0, b, acc0);
        acc1 = wmma_bf16(a1, b, acc1);
      }
#pragma unroll
      for (int r = 0; r < 8; ++r) {
        int m = r + 8 * (lane >> 4);
        int col = nt * 16 + (lane & 15);
        sT[m * 256 + col]        = f2bf(silu_f(acc0[r]));
        sT[(m + 16) * 256 + col] = f2bf(silu_f(acc1[r]));
      }
    }
  }
  __syncthreads();

  // ---- GEMM4: (32x256) @ Wc2 -> coord_w (32x1), waves 0..1 ----
  if (wave < 2) {
    int mt = wave;
    float bias = ((lane & 15) == 0) ? b_c2[0] : 0.0f;
    v8f acc;
#pragma unroll
    for (int r = 0; r < 8; ++r) acc[r] = bias;
#pragma unroll
    for (int kt = 0; kt < KT4; ++kt) {
      v16bf a = load_a_frag(sT + mt * 16 * 256, 256, lane, kt * 32);
      acc = wmma_bf16(a, load_b_frag(Wc2, NT4, kt, 0, lane), acc);
    }
    if ((lane & 15) == 0) {
#pragma unroll
      for (int r = 0; r < 8; ++r)
        sCw[mt * 16 + r + 8 * (lane >> 4)] = acc[r];
    }
  }
  __syncthreads();

  // ---- Phase 5: segment sums via f32 global atomics ----
#pragma unroll
  for (int i = 0; i < 8; ++i) {
    int idx = tid + i * 256;
    int m = idx >> 6;
    int j = idx & 63;
    atomicAdd(&m_i[(long long)sDst[m] * 64 + j], bf2f(sM[m * 64 + j]));
  }
  if (tid < 32) atomicAdd(&w_i[sDst[tid]], sCw[tid]);
  if (tid < 96) {
    int m = tid / 3, c = tid % 3;
    atomicAdd(&r_i[(long long)sDst[m] * 3 + c], sRel[m][c]);
  }
}

// ---------------------------------------------------------------------------
// Node kernel: 16 nodes / block, 256 threads (8 waves).
// ---------------------------------------------------------------------------

__global__ __launch_bounds__(256) void egnn_node_kernel(
    const float* __restrict__ x, const float* __restrict__ m_i,
    const float* __restrict__ w_i, const float* __restrict__ r_i,
    const float* __restrict__ b_n1, const float* __restrict__ b_n2,
    const bf16_t* __restrict__ Wn1, const bf16_t* __restrict__ Wn2,
    float* __restrict__ out) {
  __shared__ alignas(16) bf16_t sA[16 * 192]; // [node | m_i]
  __shared__ alignas(16) bf16_t sT[16 * 256];

  const int tid = threadIdx.x;
  const int lane = tid & 31;
  const int wave = tid >> 5;
  const int g0 = blockIdx.x * 16;

  // ---- Phase 0: A = [node(128) | m_i(64)] bf16 ----
  {
    int m = tid >> 4;
    int sub = tid & 15;
    long long g = g0 + m;
#pragma unroll
    for (int j = 0; j < 8; ++j) {
      int k = sub * 8 + j;
      sA[m * 192 + k] = f2bf(x[g * 131 + k]);
    }
#pragma unroll
    for (int j = 0; j < 4; ++j) {
      int k = sub * 4 + j;
      sA[m * 192 + 128 + k] = f2bf(m_i[g * 64 + k]);
    }
  }
  __syncthreads();

  // ---- n1: (16x192) @ Wn1 -> silu -> sT (16x256) ----
  {
    v16bf a[KTN1];
#pragma unroll
    for (int kt = 0; kt < KTN1; ++kt) a[kt] = load_a_frag(sA, 192, lane, kt * 32);
#pragma unroll
    for (int i = 0; i < 2; ++i) {
      int nt = wave + 8 * i;
      float bias = b_n1[nt * 16 + (lane & 15)];
      v8f acc;
#pragma unroll
      for (int r = 0; r < 8; ++r) acc[r] = bias;
#pragma unroll
      for (int kt = 0; kt < KTN1; ++kt)
        acc = wmma_bf16(a[kt], load_b_frag(Wn1, NTN1, kt, nt, lane), acc);
#pragma unroll
      for (int r = 0; r < 8; ++r) {
        int m = r + 8 * (lane >> 4);
        sT[m * 256 + nt * 16 + (lane & 15)] = f2bf(silu_f(acc[r]));
      }
    }
  }
  __syncthreads();

  // ---- n2: (16x256) @ Wn2 + residual -> out[:, 0:128] ----
  {
    int nt = wave; // 8 tiles, 1 per wave
    int n = nt * 16 + (lane & 15);
    float bias = b_n2[n];
    v8f acc;
#pragma unroll
    for (int r = 0; r < 8; ++r) acc[r] = bias;
#pragma unroll
    for (int kt = 0; kt < KTN2; ++kt) {
      v16bf a = load_a_frag(sT, 256, lane, kt * 32);
      acc = wmma_bf16(a, load_b_frag(Wn2, NTN2, kt, nt, lane), acc);
    }
#pragma unroll
    for (int r = 0; r < 8; ++r) {
      int m = r + 8 * (lane >> 4);
      long long g = g0 + m;
      out[g * 131 + n] = acc[r] + x[g * 131 + n];
    }
  }

  // ---- coords_out = coords + w_i * r_i -> out[:, 128:131] ----
  if (tid < 48) {
    int m = tid / 3, c = tid % 3;
    long long g = g0 + m;
    out[g * 131 + 128 + c] = x[g * 131 + 128 + c] + w_i[g] * r_i[g * 3 + c];
  }
}

// ---------------------------------------------------------------------------
// Launch
// ---------------------------------------------------------------------------

extern "C" void kernel_launch(void* const* d_in, const int* in_sizes, int n_in,
                              void* d_out, int out_size, void* d_ws,
                              size_t ws_size, hipStream_t stream) {
  const float* x    = (const float*)d_in[0];
  const int*   ei   = (const int*)d_in[1];
  const float* W_e1 = (const float*)d_in[2];
  const float* b_e1 = (const float*)d_in[3];
  const float* W_e2 = (const float*)d_in[4];
  const float* b_e2 = (const float*)d_in[5];
  const float* W_c1 = (const float*)d_in[6];
  const float* b_c1 = (const float*)d_in[7];
  const float* W_c2 = (const float*)d_in[8];
  const float* b_c2 = (const float*)d_in[9];
  const float* W_n1 = (const float*)d_in[10];
  const float* b_n1 = (const float*)d_in[11];
  const float* W_n2 = (const float*)d_in[12];
  const float* b_n2 = (const float*)d_in[13];
  float* out = (float*)d_out;

  // Workspace layout (all region sizes are multiples of 256B).
  char* ws = (char*)d_ws;
  float* m_i = (float*)ws;                       // N*64 f32
  float* w_i = m_i + (long long)NN * 64;         // N   f32
  float* r_i = w_i + NN;                         // N*3 f32
  char* wp = (char*)(r_i + (long long)NN * 3);
  bf16_t* We1b = (bf16_t*)wp; wp += (long long)KT1 * NT1 * 32 * 16 * 2;
  bf16_t* We2b = (bf16_t*)wp; wp += (long long)KT2 * NT2 * 32 * 16 * 2;
  bf16_t* Wc1b = (bf16_t*)wp; wp += (long long)KT3 * NT3 * 32 * 16 * 2;
  bf16_t* Wc2b = (bf16_t*)wp; wp += (long long)KT4 * NT4 * 32 * 16 * 2;
  bf16_t* Wn1b = (bf16_t*)wp; wp += (long long)KTN1 * NTN1 * 32 * 16 * 2;
  bf16_t* Wn2b = (bf16_t*)wp; wp += (long long)KTN2 * NTN2 * 32 * 16 * 2;

  // 1) zero accumulators (m_i, w_i, r_i are contiguous: N*68 floats)
  long long accN = (long long)NN * 68;
  zero_f32_kernel<<<(unsigned)((accN + 255) / 256), 256, 0, stream>>>(m_i, accN);

  // 2) pack weights into WMMA-B fragment layout (bf16, zero-padded)
  pack_weight_kernel<<<(KT1 * NT1 * 32 + 255) / 256, 256, 0, stream>>>(
      W_e1, We1b, K1, N1, KT1, NT1);
  pack_weight_kernel<<<(KT2 * NT2 * 32 + 255) / 256, 256, 0, stream>>>(
      W_e2, We2b, K2, N2, KT2, NT2);
  pack_weight_kernel<<<(KT3 * NT3 * 32 + 255) / 256, 256, 0, stream>>>(
      W_c1, Wc1b, K3, N3, KT3, NT3);
  pack_weight_kernel<<<(KT4 * NT4 * 32 + 255) / 256, 256, 0, stream>>>(
      W_c2, Wc2b, K4, N4, KT4, NT4);
  pack_weight_kernel<<<(KTN1 * NTN1 * 32 + 255) / 256, 256, 0, stream>>>(
      W_n1, Wn1b, KN1, NN1, KTN1, NTN1);
  pack_weight_kernel<<<(KTN2 * NTN2 * 32 + 255) / 256, 256, 0, stream>>>(
      W_n2, Wn2b, KN2, NN2, KTN2, NTN2);

  // 3) edge MLP + segment-sum scatter (32 edges per block)
  egnn_edge_kernel<<<EE / 32, 256, 0, stream>>>(
      x, ei, b_e1, b_e2, b_c1, b_c2, We1b, We2b, Wc1b, Wc2b, m_i, w_i, r_i);

  // 4) node MLP + coord update
  egnn_node_kernel<<<NN / 16, 256, 0, stream>>>(
      x, m_i, w_i, r_i, b_n1, b_n2, Wn1b, Wn2b, out);
}
